// Model_47751446397324
// MI455X (gfx1250) — matrix operand order repaired and checked by reading, not compiled, not run on hardware
//
#include <hip/hip_runtime.h>

#define HD 128           // hidden width
#define EPSV 1e-5f

typedef __attribute__((ext_vector_type(16))) __bf16 v16bf;
typedef __attribute__((ext_vector_type(8)))  float  v8f;
typedef __attribute__((ext_vector_type(8)))  unsigned short u16x8;
typedef __attribute__((ext_vector_type(4)))  unsigned short u16x4;

union Frag { v16bf v; u16x8 u[2]; };

__device__ __forceinline__ v8f wmma_bf16(const Frag& a, const Frag& b, v8f c) {
  // D = A(16x32 bf16) * B(32x16 bf16) + C(16x16 f32)
  return __builtin_amdgcn_wmma_f32_16x16x32_bf16(
      false, a.v, false, b.v, (short)0, c, false, false);
}

// CDNA5 async global->LDS copy (16B per lane), tracked by ASYNCcnt.
__device__ __forceinline__ void async_copy_b128(unsigned lds_off, const void* gptr) {
  asm volatile("global_load_async_to_lds_b128 %0, %1, off"
               :: "v"(lds_off), "v"((unsigned long long)(size_t)gptr)
               : "memory");
}
__device__ __forceinline__ void wait_asynccnt0() {
  asm volatile("s_wait_asynccnt 0x0" ::: "memory");
}
// Flat address of an LDS object: addr[31:0] is the LDS byte offset (ISA 10.2 aperture map).
__device__ __forceinline__ unsigned lds_offset(const void* p) {
  return (unsigned)(size_t)p;
}

// ---------------- elementwise / prep kernels ----------------

__global__ void zero_f32(float* p, long long n) {
  long long i = (long long)blockIdx.x * blockDim.x + threadIdx.x;
  if (i < n) p[i] = 0.f;
}

__global__ void f32_to_bf16(const float* in, __bf16* out, long long n) {
  long long i = (long long)blockIdx.x * blockDim.x + threadIdx.x;
  if (i < n) out[i] = (__bf16)in[i];
}

// Pack row-major W[Krows x 128] (f32) into WMMA-B lane layout (bf16),
// padding K up to Ksteps*32 with zeros.
// out[((c*Ksteps+s)*32 + lane)*16 + j] = W[32*s + (lane>=16?8:0) + j + (j>=8?8:0)][16*c + lane%16]
__global__ void pack_w(const float* W, __bf16* out, int Krows, int Ksteps) {
  int i = blockIdx.x * blockDim.x + threadIdx.x;
  int total = 8 * Ksteps * 32 * 16;
  if (i >= total) return;
  int j    = i & 15;
  int lane = (i >> 4) & 31;
  int cs   = i >> 9;
  int s = cs % Ksteps, c = cs / Ksteps;
  int n = 16 * c + (lane & 15);
  int k = 32 * s + ((lane >= 16) ? 8 : 0) + j + ((j >= 8) ? 8 : 0);
  float v = (k < Krows) ? W[(long long)k * HD + n] : 0.f;
  out[i] = (__bf16)v;
}

__global__ void deg_cnt(const int* ei, float* deg, float* cnt, int E) {
  int e = blockIdx.x * blockDim.x + threadIdx.x;
  if (e >= E) return;
  atomicAdd(&deg[ei[e]], 1.f);        // row degree
  atomicAdd(&cnt[ei[E + e]], 1.f);    // col count (scatter-mean)
}

__global__ void dis_inv(const float* deg, const float* cnt, float* dis, float* invc, int N) {
  int i = blockIdx.x * blockDim.x + threadIdx.x;
  if (i >= N) return;
  float d = deg[i];
  dis[i]  = (d > 0.f) ? rsqrtf(fmaxf(d, 1.f)) : 0.f;
  invc[i] = 1.f / fmaxf(cnt[i], 1.f);
}

__global__ void mean_bf(const float* hsum, const float* invc, __bf16* out, long long n) {
  long long i = (long long)blockIdx.x * blockDim.x + threadIdx.x;
  if (i < n) out[i] = (__bf16)(hsum[i] * invc[i >> 7]);
}

// ---------------- fused edge linear: gather -> WMMA -> atomic scatter ----------------
// One wave handles 16 edges x all 128 outputs. K = 160 = 4x32 (x features) + 32 (edge_attr||0).
// Packed weights (40KB) staged once per workgroup into LDS via async copy.
__global__ __launch_bounds__(256) void edge_gemm(
    const __bf16* __restrict__ xbf, const __bf16* __restrict__ eabf,
    const int* __restrict__ ei, const __bf16* __restrict__ wp,
    const float* __restrict__ bint, float* __restrict__ hsum, int E) {
  __shared__ __attribute__((aligned(16))) __bf16 Bs[8 * 5 * 32 * 16];  // 40 KB
  {
    unsigned t = threadIdx.x;
    unsigned base = lds_offset(Bs);
    const char* wb = (const char*)wp;
    #pragma unroll
    for (int it = 0; it < 10; ++it) {               // 10*256*16B = 40960B
      unsigned off = (it * 256u + t) * 16u;
      async_copy_b128(base + off, wb + off);
    }
    wait_asynccnt0();
  }
  __syncthreads();

  int wave = (int)((blockIdx.x * blockDim.x + threadIdx.x) >> 5);
  int lane = threadIdx.x & 31;
  int ebase = wave * 16;
  if (ebase >= E) return;
  const int* rowI = ei;
  const int* colI = ei + E;
  bool hi = lane >= 16;
  int colL = lane & 15;
  int myE = ebase + colL; if (myE >= E) myE = E - 1;
  int node = rowI[myE];

  v8f acc[8] = {};
  const u16x8 z8 = {0,0,0,0,0,0,0,0};
  for (int s = 0; s < 5; ++s) {
    Frag a;
    if (s < 4) {
      const __bf16* ap = xbf + (long long)node * HD + 32 * s + (hi ? 8 : 0);
      a.u[0] = *(const u16x8*)ap;
      a.u[1] = *(const u16x8*)(ap + 16);
      if (s < 3) __builtin_prefetch(ap + 32, 0, 3);
    } else {
      const __bf16* ap = eabf + (long long)myE * 16 + (hi ? 8 : 0);
      a.u[0] = *(const u16x8*)ap;    // K 128..143 = edge_attr
      a.u[1] = z8;                   // K 144..159 = pad
    }
    for (int c = 0; c < 8; ++c) {
      Frag b;
      const __bf16* bp = &Bs[((c * 5 + s) * 32 + lane) * 16];
      b.u[0] = *(const u16x8*)bp;                    // ds_load_b128
      b.u[1] = *(const u16x8*)(bp + 8);
      acc[c] = wmma_bf16(a, b, acc[c]);
    }
  }
  // C layout: lane holds (row = v + 8*(lane>=16), col = 16*c + lane%16)
  for (int c = 0; c < 8; ++c) {
    float bias = bint[c * 16 + colL];
    for (int v = 0; v < 8; ++v) {
      int er = ebase + v + (hi ? 8 : 0);
      if (er < E) {
        int n = colI[er];
        atomicAdd(&hsum[(long long)n * HD + c * 16 + colL], acc[c][v] + bias);
      }
    }
  }
}

// ---------------- node GEMM: Out[N,128] (+= Cin) A[N,128]bf16 @ Wpacked ----------------
// Packed weights (32KB) staged once per workgroup into LDS via async copy.
__global__ __launch_bounds__(256) void node_gemm(
    const __bf16* __restrict__ A, const __bf16* __restrict__ wp,
    const float* __restrict__ Cin, float* __restrict__ Out, int nrows) {
  __shared__ __attribute__((aligned(16))) __bf16 Bs[8 * 4 * 32 * 16];  // 32 KB
  {
    unsigned t = threadIdx.x;
    unsigned base = lds_offset(Bs);
    const char* wb = (const char*)wp;
    #pragma unroll
    for (int it = 0; it < 8; ++it) {                // 8*256*16B = 32768B
      unsigned off = (it * 256u + t) * 16u;
      async_copy_b128(base + off, wb + off);
    }
    wait_asynccnt0();
  }
  __syncthreads();

  int wave = (int)((blockIdx.x * blockDim.x + threadIdx.x) >> 5);
  int lane = threadIdx.x & 31;
  int rbase = wave * 16;
  if (rbase >= nrows) return;
  bool hi = lane >= 16;
  int colL = lane & 15;
  int myR = rbase + colL; if (myR >= nrows) myR = nrows - 1;

  v8f acc[8] = {};
  if (Cin) {
    for (int c = 0; c < 8; ++c)
      for (int v = 0; v < 8; ++v)
        acc[c][v] = Cin[(long long)(rbase + v + (hi ? 8 : 0)) * HD + c * 16 + colL];
  }
  for (int s = 0; s < 4; ++s) {
    Frag a;
    const __bf16* ap = A + (long long)myR * HD + 32 * s + (hi ? 8 : 0);
    a.u[0] = *(const u16x8*)ap;
    a.u[1] = *(const u16x8*)(ap + 16);
    for (int c = 0; c < 8; ++c) {
      Frag b;
      const __bf16* bp = &Bs[((c * 4 + s) * 32 + lane) * 16];
      b.u[0] = *(const u16x8*)bp;                    // ds_load_b128
      b.u[1] = *(const u16x8*)(bp + 8);
      acc[c] = wmma_bf16(a, b, acc[c]);
    }
  }
  for (int c = 0; c < 8; ++c)
    for (int v = 0; v < 8; ++v)
      Out[(long long)(rbase + v + (hi ? 8 : 0)) * HD + c * 16 + colL] = acc[c][v];
}

// ---------------- BN ----------------
__global__ void col_reduce(const float* __restrict__ Z, float* sums, int nrows) {
  int c = threadIdx.x;                 // 0..127
  float s = 0.f, q = 0.f;
  for (int r = blockIdx.x; r < nrows; r += gridDim.x) {
    float v = Z[(long long)r * HD + c];
    s += v; q += v * v;
  }
  atomicAdd(&sums[c], s);
  atomicAdd(&sums[128 + c], q);
}

__global__ void bn_relu_bf(const float* __restrict__ Z, const float* __restrict__ sums,
                           const float* __restrict__ g, const float* __restrict__ be,
                           __bf16* out, int nrows) {
  long long i = (long long)blockIdx.x * blockDim.x + threadIdx.x;
  if (i >= (long long)nrows * HD) return;
  int c = (int)(i & (HD - 1));
  float invN = 1.f / (float)nrows;
  float m = sums[c] * invN;
  float var = sums[128 + c] * invN - m * m;
  float y = g[c] * (Z[i] - m) * rsqrtf(var + EPSV) + be[c];
  out[i] = (__bf16)fmaxf(y, 0.f);
}

// ---------------- Chebyshev pieces ----------------
// acc[col] += norm_e * T[row]  (wave per edge, 4 feats/lane; all L2-resident)
__global__ __launch_bounds__(256) void lhat_scatter(
    const __bf16* __restrict__ T, const int* __restrict__ ei,
    const float* __restrict__ dis, float* __restrict__ acc, int E) {
  int wave = (int)((blockIdx.x * blockDim.x + threadIdx.x) >> 5);
  if (wave >= E) return;
  int lane = threadIdx.x & 31;
  int r = ei[wave], c = ei[E + wave];
  float nrm = -dis[r] * dis[c];
  union { u16x4 u; __bf16 b[4]; } t;
  t.u = *(const u16x4*)(T + (long long)r * HD + lane * 4);
  float* dst = acc + (long long)c * HD + lane * 4;
  #pragma unroll
  for (int j = 0; j < 4; ++j) atomicAdd(dst + j, nrm * (float)t.b[j]);
}

__global__ void tx2_combine(const float* acc2, const __bf16* tx0, __bf16* tx2, long long n) {
  long long i = (long long)blockIdx.x * blockDim.x + threadIdx.x;
  if (i < n) tx2[i] = (__bf16)(2.f * acc2[i] - (float)tx0[i]);
}

__global__ void bias_relu_bf(const float* z, const float* b, __bf16* out, long long n) {
  long long i = (long long)blockIdx.x * blockDim.x + threadIdx.x;
  if (i >= n) return;
  int c = (int)(i & (HD - 1));
  out[i] = (__bf16)fmaxf(z[i] + b[c], 0.f);
}

// ---------------- output projection H -> 1 (wave per node) ----------------
__global__ __launch_bounds__(256) void out_proj(const __bf16* __restrict__ Hb,
                                                const float* __restrict__ w,
                                                const float* __restrict__ b,
                                                float* out, int N) {
  int wave = (int)((blockIdx.x * blockDim.x + threadIdx.x) >> 5);
  if (wave >= N) return;
  int lane = threadIdx.x & 31;
  const __bf16* hp = Hb + (long long)wave * HD + lane * 4;
  float s = 0.f;
  #pragma unroll
  for (int j = 0; j < 4; ++j) s += (float)hp[j] * w[lane * 4 + j];
  for (int o = 16; o > 0; o >>= 1) s += __shfl_down(s, o, 32);
  if (lane == 0) out[wave] = s + b[0];
}

// ---------------- orchestration ----------------
extern "C" void kernel_launch(void* const* d_in, const int* in_sizes, int n_in,
                              void* d_out, int out_size, void* d_ws, size_t ws_size,
                              hipStream_t stream) {
  const float* x      = (const float*)d_in[0];
  const int*   ei     = (const int*)  d_in[1];
  const float* ea     = (const float*)d_in[2];
  const float* w_int  = (const float*)d_in[3];
  const float* b_int  = (const float*)d_in[4];
  const float* w_pre  = (const float*)d_in[5];
  // d_in[6] b_pre: cancels through BN (and is zero)
  const float* g_pre  = (const float*)d_in[7];
  const float* be_pre = (const float*)d_in[8];
  const float* c1w    = (const float*)d_in[9];
  const float* c1b    = (const float*)d_in[10];
  const float* c2w    = (const float*)d_in[11];
  const float* c2b    = (const float*)d_in[12];
  const float* w_post = (const float*)d_in[13];
  // d_in[14] b_post: cancels through BN
  const float* g_post = (const float*)d_in[15];
  const float* be_post= (const float*)d_in[16];
  const float* w_out  = (const float*)d_in[17];
  const float* b_out  = (const float*)d_in[18];

  const int N = in_sizes[0] / HD;
  const int E = in_sizes[1] / 2;
  const long long NH = (long long)N * HD;

  // ---- workspace carving ----
  char* p = (char*)d_ws;
  auto carve = [&](size_t bytes) { void* r = (void*)p; p += (bytes + 255) & ~(size_t)255; return r; };
  float*  hsum  = (float*) carve(NH * 4);          // also reused as accB
  float*  accA  = (float*) carve(NH * 4);
  float*  zbuf  = (float*) carve(NH * 4);
  __bf16* xbf   = (__bf16*)carve(NH * 2);
  __bf16* hbf0  = (__bf16*)carve(NH * 2);
  __bf16* hbf1  = (__bf16*)carve(NH * 2);
  __bf16* tx1   = (__bf16*)carve(NH * 2);
  __bf16* tx2   = (__bf16*)carve(NH * 2);
  __bf16* eabf  = (__bf16*)carve((size_t)E * 16 * 2);
  __bf16* wintp = (__bf16*)carve(8 * 5 * 32 * 16 * 2);
  __bf16* wprep = (__bf16*)carve(8 * 4 * 32 * 16 * 2);
  __bf16* c1p0  = (__bf16*)carve(8 * 4 * 32 * 16 * 2);
  __bf16* c1p1  = (__bf16*)carve(8 * 4 * 32 * 16 * 2);
  __bf16* c1p2  = (__bf16*)carve(8 * 4 * 32 * 16 * 2);
  __bf16* c2p0  = (__bf16*)carve(8 * 4 * 32 * 16 * 2);
  __bf16* c2p1  = (__bf16*)carve(8 * 4 * 32 * 16 * 2);
  __bf16* c2p2  = (__bf16*)carve(8 * 4 * 32 * 16 * 2);
  __bf16* wpostp= (__bf16*)carve(8 * 4 * 32 * 16 * 2);
  float*  degv  = (float*) carve((size_t)N * 4);
  float*  cntv  = (float*) carve((size_t)N * 4);
  float*  disv  = (float*) carve((size_t)N * 4);
  float*  invc  = (float*) carve((size_t)N * 4);
  float*  sums  = (float*) carve(256 * 4);
  (void)ws_size; (void)n_in; (void)out_size;

  auto cdiv = [](long long a, long long b) { return (unsigned)((a + b - 1) / b); };
  const int T = 256;
  unsigned gNH  = cdiv(NH, T);
  unsigned gGmm = cdiv((long long)cdiv(N, 16) * 32, T);   // node GEMM waves
  unsigned gEg  = cdiv((long long)cdiv(E, 16) * 32, T);   // edge GEMM waves
  unsigned gLh  = cdiv((long long)E * 32, T);             // lhat waves
  unsigned gOut = cdiv((long long)N * 32, T);

  // ---- prep: zero, convert, pack ----
  zero_f32<<<cdiv(N, T), T, 0, stream>>>(degv, N);
  zero_f32<<<cdiv(N, T), T, 0, stream>>>(cntv, N);
  zero_f32<<<gNH, T, 0, stream>>>(hsum, NH);
  f32_to_bf16<<<gNH, T, 0, stream>>>(x, xbf, NH);
  f32_to_bf16<<<cdiv((long long)E * 16, T), T, 0, stream>>>(ea, eabf, (long long)E * 16);
  pack_w<<<cdiv(8 * 5 * 32 * 16, T), T, 0, stream>>>(w_int, wintp, HD + 16, 5);
  pack_w<<<cdiv(8 * 4 * 32 * 16, T), T, 0, stream>>>(w_pre, wprep, HD, 4);
  pack_w<<<cdiv(8 * 4 * 32 * 16, T), T, 0, stream>>>(c1w + 0 * HD * HD, c1p0, HD, 4);
  pack_w<<<cdiv(8 * 4 * 32 * 16, T), T, 0, stream>>>(c1w + 1 * HD * HD, c1p1, HD, 4);
  pack_w<<<cdiv(8 * 4 * 32 * 16, T), T, 0, stream>>>(c1w + 2 * HD * HD, c1p2, HD, 4);
  pack_w<<<cdiv(8 * 4 * 32 * 16, T), T, 0, stream>>>(c2w + 0 * HD * HD, c2p0, HD, 4);
  pack_w<<<cdiv(8 * 4 * 32 * 16, T), T, 0, stream>>>(c2w + 1 * HD * HD, c2p1, HD, 4);
  pack_w<<<cdiv(8 * 4 * 32 * 16, T), T, 0, stream>>>(c2w + 2 * HD * HD, c2p2, HD, 4);
  pack_w<<<cdiv(8 * 4 * 32 * 16, T), T, 0, stream>>>(w_post, wpostp, HD, 4);

  // ---- degrees / normalizers ----
  deg_cnt<<<cdiv(E, T), T, 0, stream>>>(ei, degv, cntv, E);
  dis_inv<<<cdiv(N, T), T, 0, stream>>>(degv, cntv, disv, invc, N);

  // ---- edge interaction (WMMA) + scatter-mean ----
  edge_gemm<<<gEg, T, 0, stream>>>(xbf, eabf, ei, wintp, b_int, hsum, E);
  mean_bf<<<gNH, T, 0, stream>>>(hsum, invc, hbf0, NH);

  // ---- preprocess: Linear + BN + ReLU ----
  node_gemm<<<gGmm, T, 0, stream>>>(hbf0, wprep, nullptr, zbuf, N);
  zero_f32<<<1, 256, 0, stream>>>(sums, 256);
  col_reduce<<<512, 128, 0, stream>>>(zbuf, sums, N);
  bn_relu_bf<<<gNH, T, 0, stream>>>(zbuf, sums, g_pre, be_pre, hbf1, N);

  // ---- ChebConv 1: Tx0 = hbf1 ----
  zero_f32<<<gNH, T, 0, stream>>>(accA, NH);
  lhat_scatter<<<gLh, T, 0, stream>>>(hbf1, ei, disv, accA, E);
  f32_to_bf16<<<gNH, T, 0, stream>>>(accA, tx1, NH);
  zero_f32<<<gNH, T, 0, stream>>>(hsum, NH);                 // accB
  lhat_scatter<<<gLh, T, 0, stream>>>(tx1, ei, disv, hsum, E);
  tx2_combine<<<gNH, T, 0, stream>>>(hsum, hbf1, tx2, NH);
  node_gemm<<<gGmm, T, 0, stream>>>(hbf1, c1p0, nullptr, zbuf, N);
  node_gemm<<<gGmm, T, 0, stream>>>(tx1,  c1p1, zbuf,    zbuf, N);
  node_gemm<<<gGmm, T, 0, stream>>>(tx2,  c1p2, zbuf,    zbuf, N);
  bias_relu_bf<<<gNH, T, 0, stream>>>(zbuf, c1b, hbf0, NH);

  // ---- ChebConv 2: Tx0 = hbf0 ----
  zero_f32<<<gNH, T, 0, stream>>>(accA, NH);
  lhat_scatter<<<gLh, T, 0, stream>>>(hbf0, ei, disv, accA, E);
  f32_to_bf16<<<gNH, T, 0, stream>>>(accA, tx1, NH);
  zero_f32<<<gNH, T, 0, stream>>>(hsum, NH);
  lhat_scatter<<<gLh, T, 0, stream>>>(tx1, ei, disv, hsum, E);
  tx2_combine<<<gNH, T, 0, stream>>>(hsum, hbf0, tx2, NH);
  node_gemm<<<gGmm, T, 0, stream>>>(hbf0, c2p0, nullptr, zbuf, N);
  node_gemm<<<gGmm, T, 0, stream>>>(tx1,  c2p1, zbuf,    zbuf, N);
  node_gemm<<<gGmm, T, 0, stream>>>(tx2,  c2p2, zbuf,    zbuf, N);
  bias_relu_bf<<<gNH, T, 0, stream>>>(zbuf, c2b, hbf1, NH);

  // ---- postprocess: Linear + BN + ReLU ----
  node_gemm<<<gGmm, T, 0, stream>>>(hbf1, wpostp, nullptr, zbuf, N);
  zero_f32<<<1, 256, 0, stream>>>(sums, 256);
  col_reduce<<<512, 128, 0, stream>>>(zbuf, sums, N);
  bn_relu_bf<<<gNH, T, 0, stream>>>(zbuf, sums, g_post, be_post, hbf0, N);

  // ---- output projection ----
  out_proj<<<gOut, T, 0, stream>>>(hbf0, w_out, b_out, (float*)d_out, N);
}
